// xLSTMEncoderBackbone_49460843381291
// MI455X (gfx1250) — matrix operand order repaired and checked by reading, not compiled, not run on hardware
//
#include <hip/hip_runtime.h>
#include <hip/hip_bf16.h>
#include <stdint.h>

#define B_   8
#define T_   2048
#define D_   512
#define NH_  8
#define DH_  64
#define L_   4
#define KC_  4
#define UP_  704
#define EPS_ 1e-5f
#define MTOT (B_*T_)   // 16384

typedef __bf16 bf16;
typedef __attribute__((ext_vector_type(16))) __bf16 bf16x16;
typedef __attribute__((ext_vector_type(8)))  __bf16 bf16x8;
typedef __attribute__((ext_vector_type(8)))  float  f32x8;

// ---------------- WMMA fragment helpers (contiguous k = 16h + s per lane half, wave32) -------
__device__ __forceinline__ bf16x16 load_a_frag(const bf16* base, int ld) {
  int lane = threadIdx.x & 31;
  int row = lane & 15, hi = lane >> 4;
  const bf16* p = base + row * ld + hi * 16;
  bf16x8 lo = *(const bf16x8*)(p);
  bf16x8 hv = *(const bf16x8*)(p + 8);
  bf16x16 r;
#pragma unroll
  for (int i = 0; i < 8; ++i) { r[i] = lo[i]; r[i + 8] = hv[i]; }
  return r;
}
__device__ __forceinline__ bf16x16 load_b_frag(const bf16* baseT, int ld) {
  int lane = threadIdx.x & 31;
  int n = lane & 15, hi = lane >> 4;
  return *(const bf16x16*)(baseT + n * ld + hi * 16);
}
__device__ __forceinline__ f32x8 wmma_bf16(bf16x16 a, bf16x16 b, f32x8 c) {
  return __builtin_amdgcn_wmma_f32_16x16x32_bf16(false, a, false, b,
                                                 (short)0, c, false, false);
}

// ---------------- CDNA5 async LDS staging (ASYNCcnt path, inline asm) -----
// Per ISA 10.4: each lane copies 16B from its global address to its LDS address.
// The LDS byte offset is the low 32 bits of the generic (aperture) address.
__device__ __forceinline__ void async_b128(const bf16* g, bf16* l) {
  uint32_t loff = (uint32_t)(uintptr_t)l;
  asm volatile("global_load_async_to_lds_b128 %0, %1, off"
               :: "v"(loff), "v"(g) : "memory");
}
template <int N>
__device__ __forceinline__ void wait_async() {
#if __has_builtin(__builtin_amdgcn_s_wait_asynccnt)
  __builtin_amdgcn_s_wait_asynccnt(N);
#else
  asm volatile("s_wait_asynccnt %0" :: "i"(N));
#endif
}

// ---------------- math helpers --------------------------------------------
__device__ __forceinline__ float sigmf(float x) { return 1.f / (1.f + __expf(-x)); }
__device__ __forceinline__ float logsigf(float x) {
  return fminf(x, 0.f) - log1pf(__expf(-fabsf(x)));
}
__device__ __forceinline__ float gelu_exact(float x) {
  return 0.5f * x * (1.f + erff(x * 0.70710678118654752f));
}

// ---------------- elementwise / conversion kernels ------------------------
__global__ void k_mask_copy(const float* __restrict__ x,
                            const unsigned char* __restrict__ mask,
                            float* __restrict__ xw) {
  int idx = blockIdx.x * 256 + threadIdx.x;
  int m = idx >> 9;
  xw[idx] = mask[m] ? 0.f : x[idx];
}

__global__ void k_convert_bf16(const float* __restrict__ src, bf16* __restrict__ dst) {
  int i = blockIdx.x * 256 + threadIdx.x;
  dst[i] = (bf16)src[i];
}

__global__ void k_transpose_up(const float* __restrict__ src, bf16* __restrict__ dst) {
  int o = blockIdx.x * 256 + threadIdx.x;            // l*1408*512 + j*512 + d
  int l = o / (1408 * 512);
  int r = o - l * (1408 * 512);
  int j = r >> 9, d = r & 511;
  dst[o] = (bf16)src[(l * 512 + d) * 1408 + j];
}
__global__ void k_transpose_dn(const float* __restrict__ src, bf16* __restrict__ dst) {
  int o = blockIdx.x * 256 + threadIdx.x;            // l*512*704 + d*704 + u
  int l = o / (512 * 704);
  int r = o - l * (512 * 704);
  int d = r / 704, u = r - d * 704;
  dst[o] = (bf16)src[(l * 704 + u) * 512 + d];
}

// ---------------- LayerNorm (row of 512, one block per row) ---------------
__global__ void k_layernorm(const float* __restrict__ x, const float* __restrict__ w,
                            float* __restrict__ of, bf16* __restrict__ ob) {
  __shared__ float2 red[512];
  int row = blockIdx.x, d = threadIdx.x;
  float v = x[row * D_ + d];
  red[d] = make_float2(v, v * v);
  __syncthreads();
#pragma unroll
  for (int s = 256; s > 0; s >>= 1) {
    if (d < s) {
      red[d].x += red[d + s].x;
      red[d].y += red[d + s].y;
    }
    __syncthreads();
  }
  float mu = red[0].x * (1.f / D_);
  float var = red[0].y * (1.f / D_) - mu * mu;
  float y = (v - mu) * rsqrtf(var + EPS_) * w[d];
  if (of) of[row * D_ + d] = y;
  if (ob) ob[row * D_ + d] = (bf16)y;
}

// ---------------- depthwise causal conv (K=4) + SiLU -> bf16 --------------
__global__ void k_conv_silu(const float* __restrict__ xn, const float* __restrict__ cw,
                            const float* __restrict__ cb, bf16* __restrict__ xcb) {
  int idx = blockIdx.x * 256 + threadIdx.x;
  int d = idx & 511;
  int m = idx >> 9;
  int t = m & (T_ - 1);
  float acc = cb[d];
#pragma unroll
  for (int k = 0; k < KC_; ++k) {
    int tt = t - (KC_ - 1) + k;
    if (tt >= 0) acc += cw[k * D_ + d] * xn[(m - (KC_ - 1) + k) * D_ + d];
  }
  float y = acc * sigmf(acc);
  xcb[idx] = (bf16)y;
}

// ---------------- gate input projections (WMMA, register blocked) ---------
// grid (MTOT/64, 32), block 128: wave w owns M-subtile w, all 4 N-tiles.
// gates layout: [t][hd][g][b][e]
__global__ void k_gates_gemm(const bf16* __restrict__ xc, const bf16* __restrict__ xn,
                             const bf16* __restrict__ wg, const float* __restrict__ bgl,
                             float* __restrict__ gates) {
  int wave = threadIdx.x >> 5;
  int m0 = blockIdx.x * 64 + wave * 16;
  int g = blockIdx.y >> 3, hd = blockIdx.y & 7;
  int lane = threadIdx.x & 31, nl = lane & 15, hi = lane >> 4;

  const bf16* src = (g < 2) ? xc : xn;
  const bf16* A = src + (size_t)m0 * D_ + hd * DH_;
  const bf16* BT = wg + (size_t)((g * 8 + hd) * 64) * 64;    // [e][d], ld = 64

  bf16x16 a0 = load_a_frag(A, D_);
  bf16x16 a1 = load_a_frag(A + 32, D_);

#pragma unroll
  for (int nt = 0; nt < 4; ++nt) {
    int e = nt * 16 + nl;
    float bias = bgl[(g * 8 + hd) * 64 + e];
    f32x8 c;
#pragma unroll
    for (int v = 0; v < 8; ++v) c[v] = bias;
    bf16x16 b0 = load_b_frag(BT + nt * 16 * 64, 64);
    bf16x16 b1 = load_b_frag(BT + nt * 16 * 64 + 32, 64);
    c = wmma_bf16(a0, b0, c);
    c = wmma_bf16(a1, b1, c);
#pragma unroll
    for (int v = 0; v < 8; ++v) {
      int m = m0 + v + hi * 8;
      int b = m >> 11, t = m & (T_ - 1);
      gates[(((t * 8 + hd) * 4 + g) * 8 + b) * 64 + e] = c[v];
    }
  }
}

// ---------------- sequential sLSTM scan (WMMA recurrence) -----------------
// grid 8 (one block per head), block 512 = 16 waves: wave -> (gate, n-tile)
__global__ void k_scan(const float* __restrict__ gates, const bf16* __restrict__ rt,
                       float* __restrict__ hout) {
  int hd = blockIdx.x;
  int tid = threadIdx.x;
  int wave = tid >> 5, g = wave >> 2, nt = wave & 3;
  int lane = tid & 31, nl = lane & 15, hi = lane >> 4;
  int e = nt * 16 + nl;

  __shared__ __align__(16) bf16 s_h[16 * 64];
  __shared__ float s_r[4 * 16 * 64];
  __shared__ float s_c[8 * 64], s_n[8 * 64], s_m[8 * 64];

  for (int i = tid; i < 16 * 64; i += 512) s_h[i] = (bf16)0.f;
  s_c[tid] = 0.f; s_n[tid] = 0.f; s_m[tid] = 0.f;
  __syncthreads();

  const bf16* BT = rt + (size_t)((g * 8 + hd) * 64) * 64;
  bf16x16 b0 = load_b_frag(BT + nt * 16 * 64, 64);
  bf16x16 b1 = load_b_frag(BT + nt * 16 * 64 + 32, 64);

  int brow = tid >> 6, ee = tid & 63;

  for (int t = 0; t < T_; ++t) {
    // issue global gate loads early: latency hides under WMMA + barrier
    int base = ((t * 8 + hd) * 4) * 512 + brow * 64 + ee;
    float gi = gates[base];
    float gf = gates[base + 512];
    float gz = gates[base + 1024];
    float go = gates[base + 1536];

    bf16x16 a0 = load_a_frag(s_h, 64);
    bf16x16 a1 = load_a_frag(s_h + 32, 64);
    f32x8 c = {};
    c = wmma_bf16(a0, b0, c);
    c = wmma_bf16(a1, b1, c);
#pragma unroll
    for (int v = 0; v < 8; ++v)
      s_r[(g * 16 + v + hi * 8) * 64 + e] = c[v];
    __syncthreads();

    {
      float ip = gi + s_r[( 0 + brow) * 64 + ee];
      float fp = gf + s_r[(16 + brow) * 64 + ee];
      float zp = gz + s_r[(32 + brow) * 64 + ee];
      float op = go + s_r[(48 + brow) * 64 + ee];
      float lf = s_m[tid] + logsigf(fp);
      float mn = fmaxf(ip, lf);
      float iv = __expf(ip - mn);
      float fv = __expf(lf - mn);
      float cn = fv * s_c[tid] + iv * tanhf(zp);
      float nn = fv * s_n[tid] + iv;
      float hn = sigmf(op) * cn / nn;
      s_c[tid] = cn; s_n[tid] = nn; s_m[tid] = mn;
      s_h[brow * 64 + ee] = (bf16)hn;
      hout[(brow * T_ + t) * D_ + hd * 64 + ee] = hn;
    }
    __syncthreads();
  }
}

// ---------------- per-head GroupNorm + residual add -----------------------
__global__ void k_groupnorm_res(const float* __restrict__ h, const float* __restrict__ gn,
                                float* __restrict__ xw) {
  __shared__ float sv[512];
  __shared__ float s_mu[8], s_rs[8];
  int row = blockIdx.x, d = threadIdx.x;
  float v = h[row * D_ + d];
  sv[d] = v;
  __syncthreads();
  if (d < 8) {
    float s = 0.f, s2 = 0.f;
#pragma unroll
    for (int j = 0; j < 64; ++j) { float x = sv[d * 64 + j]; s += x; s2 += x * x; }
    float mu = s * (1.f / 64.f);
    float var = s2 * (1.f / 64.f) - mu * mu;
    s_mu[d] = mu;
    s_rs[d] = rsqrtf(var + EPS_);
  }
  __syncthreads();
  int hd = d >> 6;
  xw[row * D_ + d] += (v - s_mu[hd]) * s_rs[hd] * gn[d];
}

// ---------------- register-blocked WMMA GEMM with async-LDS A staging -----
// Block tile M=64 x N=128; 8 waves (wy=wave>>2 in M, wx=wave&3 in N);
// each wave owns a 32x32 register tile (4 accumulators).
// A tile (64 x 32 per K-chunk) double-buffered in LDS via ASYNCcnt loads.
template <int LDA, int LDB, int NITER, bool ACCUM>
__device__ __forceinline__ void gemm_block(const bf16* __restrict__ A,   // m0 row 0
                                           const bf16* __restrict__ BT,  // n0 row 0
                                           float* __restrict__ out,      // [m][ldout]
                                           int m0, int n0, int ldout) {
  __shared__ __align__(16) bf16 s_a[2][64][32];
  int tid = threadIdx.x;
  int wave = tid >> 5, wy = wave >> 2, wx = wave & 3;
  int lane = tid & 31, nl = lane & 15, hi = lane >> 4;
  int srow = tid >> 2, sseg = tid & 3;               // staging: 64 rows x 4 segs

  f32x8 c[2][2] = {};

  auto compute = [&](int kc) {
    int buf = kc & 1;
    bf16x16 a0 = load_a_frag(&s_a[buf][wy * 32][0], 32);
    bf16x16 a1 = load_a_frag(&s_a[buf][wy * 32 + 16][0], 32);
    bf16x16 b0 = load_b_frag(BT + (size_t)(wx * 32) * LDB + kc * 32, LDB);
    bf16x16 b1 = load_b_frag(BT + (size_t)(wx * 32 + 16) * LDB + kc * 32, LDB);
    c[0][0] = wmma_bf16(a0, b0, c[0][0]);
    c[0][1] = wmma_bf16(a0, b1, c[0][1]);
    c[1][0] = wmma_bf16(a1, b0, c[1][0]);
    c[1][1] = wmma_bf16(a1, b1, c[1][1]);
  };

  // prologue: stage K-chunk 0
  async_b128(A + (size_t)srow * LDA + sseg * 8, &s_a[0][srow][sseg * 8]);

#pragma unroll 1
  for (int kc = 0; kc < NITER - 1; ++kc) {
    async_b128(A + (size_t)srow * LDA + (kc + 1) * 32 + sseg * 8,
               &s_a[(kc + 1) & 1][srow][sseg * 8]);
    wait_async<1>();                                 // chunk kc landed (in-order)
    __syncthreads();
    compute(kc);
    __syncthreads();
  }
  wait_async<0>();
  __syncthreads();
  compute(NITER - 1);

#pragma unroll
  for (int mf = 0; mf < 2; ++mf)
#pragma unroll
    for (int nf = 0; nf < 2; ++nf)
#pragma unroll
      for (int v = 0; v < 8; ++v) {
        size_t idx = (size_t)(m0 + wy * 32 + mf * 16 + v + hi * 8) * ldout +
                     (n0 + wx * 32 + nf * 16 + nl);
        if (ACCUM) out[idx] += c[mf][nf][v];
        else       out[idx]  = c[mf][nf][v];
      }
}

// FFN up: M=16384, K=512, N=1408.  grid (256, 11), block 256
__global__ void k_ffn1(const bf16* __restrict__ xln, const bf16* __restrict__ wupT,
                       float* __restrict__ u) {
  int m0 = blockIdx.x * 64, n0 = blockIdx.y * 128;
  gemm_block<512, 512, 16, false>(xln + (size_t)m0 * 512, wupT + (size_t)n0 * 512,
                                  u, m0, n0, 2 * UP_);
}

// FFN down + residual: M=16384, K=704, N=512.  grid (256, 4), block 256
__global__ void k_ffn2(const bf16* __restrict__ vb, const bf16* __restrict__ wdnT,
                       float* __restrict__ xw) {
  int m0 = blockIdx.x * 64, n0 = blockIdx.y * 128;
  gemm_block<704, 704, 22, true>(vb + (size_t)m0 * 704, wdnT + (size_t)n0 * 704,
                                 xw, m0, n0, D_);
}

// ---------------- GeGLU activation -> bf16 --------------------------------
__global__ void k_geglu(const float* __restrict__ u, bf16* __restrict__ vb) {
  int idx = blockIdx.x * 256 + threadIdx.x;
  int m = idx / UP_, j = idx - m * UP_;
  float gt = u[(size_t)m * (2 * UP_) + j];
  float up = u[(size_t)m * (2 * UP_) + UP_ + j];
  vb[idx] = (bf16)(gelu_exact(gt) * up);
}

// ---------------- host-side orchestration ---------------------------------
extern "C" void kernel_launch(void* const* d_in, const int* in_sizes, int n_in,
                              void* d_out, int out_size, void* d_ws, size_t ws_size,
                              hipStream_t stream) {
  (void)in_sizes; (void)n_in; (void)out_size; (void)ws_size;
  const float* x_in    = (const float*)d_in[0];
  const unsigned char* mask = (const unsigned char*)d_in[1];
  const float* ln1_w   = (const float*)d_in[2];
  const float* conv_w  = (const float*)d_in[3];
  const float* conv_b  = (const float*)d_in[4];
  const float* Wg      = (const float*)d_in[5];
  const float* Rg      = (const float*)d_in[6];
  const float* bg      = (const float*)d_in[7];
  const float* gn_w    = (const float*)d_in[8];
  const float* ln2_w   = (const float*)d_in[9];
  const float* w_up    = (const float*)d_in[10];
  const float* w_down  = (const float*)d_in[11];
  const float* post_ln = (const float*)d_in[12];
  float* out = (float*)d_out;

  uint8_t* p = (uint8_t*)d_ws;
  auto alloc = [&](size_t bytes) -> void* {
    void* q = (void*)p;
    p += (bytes + 255) & ~(size_t)255;
    return q;
  };
  float* XW   = (float*)alloc((size_t)MTOT * D_ * 4);
  float* XN   = (float*)alloc((size_t)MTOT * D_ * 4);
  bf16*  XNB  = (bf16*) alloc((size_t)MTOT * D_ * 2);
  bf16*  XCB  = (bf16*) alloc((size_t)MTOT * D_ * 2);
  float* H    = (float*)alloc((size_t)MTOT * D_ * 4);
  bf16*  XLNB = (bf16*) alloc((size_t)MTOT * D_ * 2);
  float* GU   = (float*)alloc((size_t)4 * MTOT * D_ * 4);    // gates / FFN-u union
  bf16*  VB   = (bf16*) alloc((size_t)MTOT * UP_ * 2);
  bf16*  WGB  = (bf16*) alloc((size_t)L_ * 4 * NH_ * DH_ * DH_ * 2);
  bf16*  RGB  = (bf16*) alloc((size_t)L_ * 4 * NH_ * DH_ * DH_ * 2);
  bf16*  WUPT = (bf16*) alloc((size_t)L_ * 2 * UP_ * D_ * 2);
  bf16*  WDNT = (bf16*) alloc((size_t)L_ * D_ * UP_ * 2);

  k_convert_bf16<<<(L_ * 4 * NH_ * DH_ * DH_) / 256, 256, 0, stream>>>(Wg, WGB);
  k_convert_bf16<<<(L_ * 4 * NH_ * DH_ * DH_) / 256, 256, 0, stream>>>(Rg, RGB);
  k_transpose_up<<<(L_ * 2 * UP_ * D_) / 256, 256, 0, stream>>>(w_up, WUPT);
  k_transpose_dn<<<(L_ * D_ * UP_) / 256, 256, 0, stream>>>(w_down, WDNT);

  k_mask_copy<<<(MTOT * D_) / 256, 256, 0, stream>>>(x_in, mask, XW);

  for (int l = 0; l < L_; ++l) {
    const float* ln1l = ln1_w + l * D_;
    const float* cwl  = conv_w + l * KC_ * D_;
    const float* cbl  = conv_b + l * D_;
    const bf16*  wgl  = WGB + (size_t)l * 4 * NH_ * DH_ * DH_;
    const bf16*  rgl  = RGB + (size_t)l * 4 * NH_ * DH_ * DH_;
    const float* bgl  = bg + (size_t)l * 4 * NH_ * DH_;
    const float* gnl  = gn_w + l * D_;
    const float* ln2l = ln2_w + l * D_;
    const bf16*  wupl = WUPT + (size_t)l * 2 * UP_ * D_;
    const bf16*  wdnl = WDNT + (size_t)l * D_ * UP_;

    k_layernorm<<<MTOT, 512, 0, stream>>>(XW, ln1l, XN, XNB);
    k_conv_silu<<<(MTOT * D_) / 256, 256, 0, stream>>>(XN, cwl, cbl, XCB);
    k_gates_gemm<<<dim3(MTOT / 64, 32), 128, 0, stream>>>(XCB, XNB, wgl, bgl, GU);
    k_scan<<<NH_, 512, 0, stream>>>(GU, rgl, H);
    k_groupnorm_res<<<MTOT, 512, 0, stream>>>(H, gnl, XW);
    k_layernorm<<<MTOT, 512, 0, stream>>>(XW, ln2l, nullptr, XLNB);
    k_ffn1<<<dim3(MTOT / 64, (2 * UP_) / 128), 256, 0, stream>>>(XLNB, wupl, GU);
    k_geglu<<<(MTOT * UP_) / 256, 256, 0, stream>>>(GU, VB);
    k_ffn2<<<dim3(MTOT / 64, D_ / 128), 256, 0, stream>>>(VB, wdnl, XW);
  }

  k_layernorm<<<MTOT, 512, 0, stream>>>(XW, post_ln, out, nullptr);
}
